// SWinHGNNet_30305289240775
// MI455X (gfx1250) — matrix operand order repaired and checked by reading, not compiled, not run on hardware
//
#include <hip/hip_runtime.h>
#include <cstddef>
#include <cstdint>

// ---------------------------------------------------------------------------
// Problem constants (match reference)
// ---------------------------------------------------------------------------
#define BATCH 8
#define N0    2000
#define N1    500
#define N2    125
#define D_IN  1024
#define H_UP  512
#define HG    512
#define TGT   128
#define KNN_K 10
#define NEG_SLOPE 0.01f

// ---------------------------------------------------------------------------
// Types for CDNA5 WMMA (wave32, 16x16x32 bf16 -> f32)
// ---------------------------------------------------------------------------
typedef __attribute__((ext_vector_type(16))) __bf16     v16bf;
typedef __attribute__((ext_vector_type(8)))  float      v8f;
typedef __attribute__((ext_vector_type(4)))  unsigned   v4u;

union Frag32B { v16bf v; v4u u[2]; };   // 32-byte bf16 fragment (16 elems)

// float -> bf16 (round to nearest even), bf16 -> float helpers on raw bits
__device__ __forceinline__ unsigned short f2bf(float f) {
    unsigned u = __float_as_uint(f);
    u += 0x7fffu + ((u >> 16) & 1u);
    return (unsigned short)(u >> 16);
}
__device__ __forceinline__ float bf2f(unsigned short h) {
    return __uint_as_float(((unsigned)h) << 16);
}

// ---------------------------------------------------------------------------
// Weight transpose + bf16 convert:  Wt[n*K + k] = bf16(W[k*N + n])
// ---------------------------------------------------------------------------
__global__ void k_transpose_bf16(const float* __restrict__ W,
                                 unsigned short* __restrict__ Wt,
                                 int K, int N) {
    int idx = blockIdx.x * blockDim.x + threadIdx.x;
    if (idx >= K * N) return;
    int n = idx / K, k = idx % K;
    Wt[(size_t)n * K + k] = f2bf(W[(size_t)k * N + n]);
}

// Plain f32 -> bf16 convert
__global__ void k_f32_to_bf16(const float* __restrict__ src,
                              unsigned short* __restrict__ dst, int n) {
    int idx = blockIdx.x * blockDim.x + threadIdx.x;
    if (idx < n) dst[idx] = f2bf(src[idx]);
}

// Row norms (sum of squares) of a bf16 matrix [rows x Kd]
__global__ void k_rownorms(const unsigned short* __restrict__ X,
                           float* __restrict__ norms, int Kd) {
    __shared__ float red[128];
    int row = blockIdx.x;
    const unsigned short* xr = X + (size_t)row * Kd;
    float s = 0.f;
    for (int k = threadIdx.x; k < Kd; k += 128) {
        float v = bf2f(xr[k]);
        s += v * v;
    }
    red[threadIdx.x] = s;
    __syncthreads();
    for (int off = 64; off > 0; off >>= 1) {
        if (threadIdx.x < off) red[threadIdx.x] += red[threadIdx.x + off];
        __syncthreads();
    }
    if (threadIdx.x == 0) norms[row] = red[0];
}

// ---------------------------------------------------------------------------
// Fused kNN: per (batch, 16-row tile), Gram via WMMA over all 2000 columns in
// groups of 5x16 cols, rank by score = norm[n] - 2*dot, keep top-10 per row.
// One wave per block. X: [BATCH, N, Kd] bf16 row-major.
// ---------------------------------------------------------------------------
#define KNN_NT 5   // col tiles per A-fragment reuse; 2000/16 = 125 = 25 * 5

__global__ __launch_bounds__(32)
void k_knn(const unsigned short* __restrict__ X,
           const float* __restrict__ norms,
           int* __restrict__ nn, int N, int Kd) {
    int tiles_per_b = N / 16;
    int b  = blockIdx.x / tiles_per_b;
    int rt = blockIdx.x % tiles_per_b;

    const unsigned short* Xb = X + (size_t)b * N * Kd;
    const float* nb = norms + (size_t)b * N;

    int lane = threadIdx.x;
    int half = lane >> 4;      // 0: low half-wave, 1: high half-wave
    int low  = lane & 15;
    int arow = rt * 16 + low;  // A-matrix row handled by this lane

    // per-row top-K (only lanes 0..15 own a row)
    float best[KNN_K];
    int   bidx[KNN_K];
#pragma unroll
    for (int j = 0; j < KNN_K; ++j) { best[j] = 3.4e38f; bidx[j] = 0; }
    float worst = 3.4e38f; int wp = 0;

    __shared__ float sc[16][16];

    const int ngroups = N / (16 * KNN_NT);
    for (int cg = 0; cg < ngroups; ++cg) {
        v8f acc[KNN_NT] = {};
        for (int k0 = 0; k0 < Kd; k0 += 32) {
            // A fragment (16x32 bf16): lane half h -> K = k0+8h..+7 and k0+16+8h..+7
            Frag32B a;
            const unsigned short* ap = Xb + (size_t)arow * Kd + k0 + half * 8;
            a.u[0] = *(const v4u*)(ap);
            a.u[1] = *(const v4u*)(ap + 16);
#pragma unroll
            for (int t = 0; t < KNN_NT; ++t) {
                // B fragment (32x16 bf16): lane half h -> K = k0+8h..+7 and k0+16+8h..+7, N = low
                int col = (cg * KNN_NT + t) * 16 + low;
                Frag32B bf;
                const unsigned short* bp = Xb + (size_t)col * Kd + k0 + half * 8;
                bf.u[0] = *(const v4u*)(bp);
                bf.u[1] = *(const v4u*)(bp + 16);
                acc[t] = __builtin_amdgcn_wmma_f32_16x16x32_bf16(
                    false, a.v, false, bf.v, (short)0, acc[t], false, false);
            }
        }
        // score tiles -> LDS -> per-row top-K update
#pragma unroll
        for (int t = 0; t < KNN_NT; ++t) {
            int cbase = (cg * KNN_NT + t) * 16;
#pragma unroll
            for (int r = 0; r < 8; ++r) {
                int m = half * 8 + r;     // C/D layout: VGPR r -> M = r + 8*half
                sc[m][low] = nb[cbase + low] - 2.0f * acc[t][r];
            }
            __syncthreads();
            if (lane < 16) {
                for (int n = 0; n < 16; ++n) {
                    float s = sc[lane][n];
                    if (s < worst) {
                        int ci = cbase + n;
#pragma unroll
                        for (int j = 0; j < KNN_K; ++j)
                            if (j == wp) { best[j] = s; bidx[j] = ci; }
                        worst = -3.4e38f;
#pragma unroll
                        for (int j = 0; j < KNN_K; ++j)
                            if (best[j] > worst) { worst = best[j]; wp = j; }
                    }
                }
            }
            __syncthreads();
        }
    }
    if (lane < 16) {
        int e = rt * 16 + lane;
        int* out = nn + ((size_t)b * N + e) * KNN_K;
#pragma unroll
        for (int j = 0; j < KNN_K; ++j) out[j] = bidx[j];
    }
}

// ---------------------------------------------------------------------------
// WMMA GEMM: C[M,N] = A[M,Kd](bf16) x Wt[N,Kd](bf16, pre-transposed) (+bias)
// One wave per 16x16 output tile. Rows clamped, stores guarded (M % 16 != 0).
// ---------------------------------------------------------------------------
__global__ __launch_bounds__(32)
void k_gemm(const unsigned short* __restrict__ A,
            const unsigned short* __restrict__ Wt,
            const float* __restrict__ bias,
            float* __restrict__ C, int M, int N, int Kd) {
    int mt = blockIdx.x, nt = blockIdx.y;
    int lane = threadIdx.x;
    int half = lane >> 4, low = lane & 15;
    int arow = mt * 16 + low; if (arow > M - 1) arow = M - 1;
    int ncol = nt * 16 + low;

    v8f acc = {};
    for (int k0 = 0; k0 < Kd; k0 += 32) {
        Frag32B a;
        const unsigned short* ap = A + (size_t)arow * Kd + k0 + half * 8;
        a.u[0] = *(const v4u*)(ap);
        a.u[1] = *(const v4u*)(ap + 16);
        Frag32B bf;
        const unsigned short* bp = Wt + (size_t)ncol * Kd + k0 + half * 8;
        bf.u[0] = *(const v4u*)(bp);
        bf.u[1] = *(const v4u*)(bp + 16);
        acc = __builtin_amdgcn_wmma_f32_16x16x32_bf16(
            false, a.v, false, bf.v, (short)0, acc, false, false);
    }
    float bv = bias ? bias[ncol] : 0.0f;
#pragma unroll
    for (int r = 0; r < 8; ++r) {
        int m = mt * 16 + half * 8 + r;
        if (m < M) C[(size_t)m * N + ncol] = acc[r] + bv;
    }
}

// ---------------------------------------------------------------------------
// Hyperedge mean: edge[b,e,:] = (1/K) * sum_j xprime[b, nn[b,e,j], :]
// One block (256 thr) per (b,e); each thread covers dims d and d+256.
// ---------------------------------------------------------------------------
__global__ void k_edge_mean(const float* __restrict__ xp,
                            const int* __restrict__ nn,
                            float* __restrict__ edge, int N) {
    __shared__ int sidx[KNN_K];
    int be = blockIdx.x;                 // b*N + e
    int b  = be / N;
    if (threadIdx.x < KNN_K) sidx[threadIdx.x] = nn[(size_t)be * KNN_K + threadIdx.x];
    __syncthreads();
#pragma unroll
    for (int h = 0; h < 2; ++h) {
        int d = threadIdx.x + h * 256;
        float s = 0.f;
#pragma unroll
        for (int j = 0; j < KNN_K; ++j)
            s += xp[((size_t)b * N + sidx[j]) * HG + d];
        edge[(size_t)be * HG + d] = s * (1.0f / KNN_K);
    }
}

// Node degree counts Dv (appearance count of node in nn lists)
__global__ void k_count(const int* __restrict__ nn, int* __restrict__ cnt,
                        int total, int N) {
    int id = blockIdx.x * blockDim.x + threadIdx.x;
    if (id >= total) return;
    int b = id / N;
#pragma unroll
    for (int j = 0; j < KNN_K; ++j)
        atomicAdd(&cnt[b * N + nn[(size_t)id * KNN_K + j]], 1);
}

// Scatter edge features back to member nodes: acc[nn[e,j]] += edge[e]
__global__ void k_scatter(const float* __restrict__ edge,
                          const int* __restrict__ nn,
                          float* __restrict__ acc, int N) {
    __shared__ int sidx[KNN_K];
    int be = blockIdx.x;
    int b  = be / N;
    if (threadIdx.x < KNN_K) sidx[threadIdx.x] = nn[(size_t)be * KNN_K + threadIdx.x];
    __syncthreads();
#pragma unroll
    for (int h = 0; h < 2; ++h) {
        int d = threadIdx.x + h * 256;
        float v = edge[(size_t)be * HG + d];
#pragma unroll
        for (int j = 0; j < KNN_K; ++j)
            atomicAdd(&acc[((size_t)b * N + sidx[j]) * HG + d], v);
    }
}

// hidden = leaky_relu(acc / max(Dv,1) + bias); write f32 and bf16 (into hcat)
__global__ void k_finalize(const float* __restrict__ acc,
                           const int* __restrict__ cnt,
                           const float* __restrict__ bias,
                           float* __restrict__ hid,
                           unsigned short* __restrict__ hcat, int total) {
    int idx = blockIdx.x * blockDim.x + threadIdx.x;
    if (idx >= total) return;
    int row = idx / HG, d = idx % HG;
    float dv = (float)max(cnt[row], 1);
    float v = acc[idx] / dv + bias[d];
    v = (v > 0.f) ? v : NEG_SLOPE * v;
    hid[idx] = v;
    hcat[(size_t)row * D_IN + d] = f2bf(v);
}

// argmin over 2D coords: nearest[b,n0] = argmin_i ||c0[b,n0] - ci[b,i]||^2
__global__ void k_nearest(const float* __restrict__ c0,
                          const float* __restrict__ ci,
                          int* __restrict__ nearest, int Ni) {
    int id = blockIdx.x * blockDim.x + threadIdx.x;
    if (id >= BATCH * N0) return;
    int b = id / N0;
    float px = c0[(size_t)id * 2], py = c0[(size_t)id * 2 + 1];
    const float* cb = ci + (size_t)b * Ni * 2;
    float bd = 3.4e38f; int bi = 0;
    for (int i = 0; i < Ni; ++i) {
        float dx = px - cb[2 * i], dy = py - cb[2 * i + 1];
        float d = dx * dx + dy * dy;
        if (d < bd) { bd = d; bi = i; }
    }
    nearest[id] = bi;
}

// hcat[:, 512:1024] = bf16(e_i[nearest])
__global__ void k_gather(const float* __restrict__ ei,
                         const int* __restrict__ nearest,
                         unsigned short* __restrict__ hcat, int Ni, int total) {
    int idx = blockIdx.x * blockDim.x + threadIdx.x;
    if (idx >= total) return;
    int row = idx / H_UP, d = idx % H_UP;   // row = b*N0 + n0
    int b = row / N0;
    int t = nearest[row];
    hcat[(size_t)row * D_IN + H_UP + d] =
        f2bf(ei[((size_t)b * Ni + t) * H_UP + d]);
}

// mean pool over nodes
__global__ void k_pool(const float* __restrict__ hid, float* __restrict__ pool) {
    int idx = blockIdx.x * blockDim.x + threadIdx.x;
    if (idx >= BATCH * HG) return;
    int b = idx / HG, d = idx % HG;
    float s = 0.f;
    for (int n = 0; n < N0; ++n) s += hid[((size_t)b * N0 + n) * HG + d];
    pool[idx] = s * (1.0f / N0);
}

// out = pool @ Wlast + blast   (tiny: 8x512x128, plain FMA)
__global__ void k_final(const float* __restrict__ pool,
                        const float* __restrict__ Wlast,
                        const float* __restrict__ blast,
                        float* __restrict__ out) {
    int idx = blockIdx.x * blockDim.x + threadIdx.x;
    if (idx >= BATCH * TGT) return;
    int b = idx / TGT, t = idx % TGT;
    float s = blast[t];
    for (int d = 0; d < HG; ++d)
        s += pool[(size_t)b * HG + d] * Wlast[(size_t)d * TGT + t];
    out[idx] = s;
}

// ---------------------------------------------------------------------------
// Host orchestration
// ---------------------------------------------------------------------------
extern "C" void kernel_launch(void* const* d_in, const int* in_sizes, int n_in,
                              void* d_out, int out_size, void* d_ws, size_t ws_size,
                              hipStream_t stream) {
    const float* x0     = (const float*)d_in[0];
    const float* x1     = (const float*)d_in[1];
    const float* x2     = (const float*)d_in[2];
    const float* c0     = (const float*)d_in[3];
    const float* c1     = (const float*)d_in[4];
    const float* c2     = (const float*)d_in[5];
    const float* theta0 = (const float*)d_in[6];
    const float* bias0  = (const float*)d_in[7];
    const float* theta1 = (const float*)d_in[8];
    const float* bias1  = (const float*)d_in[9];
    const float* theta2 = (const float*)d_in[10];
    const float* bias2  = (const float*)d_in[11];
    const float* Wf1    = (const float*)d_in[12];
    const float* bf1    = (const float*)d_in[13];
    const float* Wf2    = (const float*)d_in[14];
    const float* bf2    = (const float*)d_in[15];
    const float* Wlast  = (const float*)d_in[16];
    const float* blast  = (const float*)d_in[17];

    // bump allocator on workspace
    char* ws = (char*)d_ws;
    size_t off = 0;
    auto alloc = [&](size_t bytes) -> void* {
        off = (off + 255) & ~(size_t)255;
        void* p = ws + off;
        off += bytes;
        return p;
    };

    const int ROWS = BATCH * N0;                       // 16000
    unsigned short* t0t  = (unsigned short*)alloc((size_t)HG * D_IN * 2);
    unsigned short* t1t  = (unsigned short*)alloc((size_t)HG * D_IN * 2);
    unsigned short* t2t  = (unsigned short*)alloc((size_t)HG * D_IN * 2);
    unsigned short* wf1t = (unsigned short*)alloc((size_t)H_UP * D_IN * 2);
    unsigned short* wf2t = (unsigned short*)alloc((size_t)H_UP * D_IN * 2);
    unsigned short* x1b  = (unsigned short*)alloc((size_t)BATCH * N1 * D_IN * 2);
    unsigned short* x2b  = (unsigned short*)alloc((size_t)BATCH * N2 * D_IN * 2);
    unsigned short* hcat = (unsigned short*)alloc((size_t)ROWS * D_IN * 2);
    float* norms   = (float*)alloc((size_t)ROWS * 4);
    int*   nn      = (int*)alloc((size_t)ROWS * KNN_K * 4);
    float* xprime  = (float*)alloc((size_t)ROWS * HG * 4);
    float* edge    = (float*)alloc((size_t)ROWS * HG * 4);
    float* accv    = (float*)alloc((size_t)ROWS * HG * 4);
    int*   cnt     = (int*)alloc((size_t)ROWS * 4);
    float* hid     = (float*)alloc((size_t)ROWS * HG * 4);
    float* e1      = (float*)alloc((size_t)BATCH * N1 * H_UP * 4);
    float* e2      = (float*)alloc((size_t)BATCH * N2 * H_UP * 4);
    int*   nearest = (int*)alloc((size_t)ROWS * 4);
    float* pool    = (float*)alloc((size_t)BATCH * HG * 4);

    // --- weight transposes + bf16 converts ---
    {
        int n = D_IN * HG;
        dim3 g((n + 255) / 256), blk(256);
        k_transpose_bf16<<<g, blk, 0, stream>>>(theta0, t0t, D_IN, HG);
        k_transpose_bf16<<<g, blk, 0, stream>>>(theta1, t1t, D_IN, HG);
        k_transpose_bf16<<<g, blk, 0, stream>>>(theta2, t2t, D_IN, HG);
        k_transpose_bf16<<<g, blk, 0, stream>>>(Wf1, wf1t, D_IN, H_UP);
        k_transpose_bf16<<<g, blk, 0, stream>>>(Wf2, wf2t, D_IN, H_UP);
    }
    {
        int n0e = ROWS * D_IN;
        k_f32_to_bf16<<<(n0e + 255) / 256, 256, 0, stream>>>(x0, hcat, n0e);
        int n1e = BATCH * N1 * D_IN;
        k_f32_to_bf16<<<(n1e + 255) / 256, 256, 0, stream>>>(x1, x1b, n1e);
        int n2e = BATCH * N2 * D_IN;
        k_f32_to_bf16<<<(n2e + 255) / 256, 256, 0, stream>>>(x2, x2b, n2e);
    }
    k_rownorms<<<ROWS, 128, 0, stream>>>(hcat, norms, D_IN);

    const unsigned short* thT[3] = { t0t, t1t, t2t };
    const float*          bsT[3] = { bias0, bias1, bias2 };

    for (int l = 0; l < 3; ++l) {
        // 1) kNN incidence (fused WMMA Gram + top-10)
        k_knn<<<BATCH * (N0 / 16), 32, 0, stream>>>(hcat, norms, nn, N0, D_IN);
        // 2) x' = hcat @ theta_l
        k_gemm<<<dim3(ROWS / 16, HG / 16), 32, 0, stream>>>(
            hcat, thT[l], nullptr, xprime, ROWS, HG, D_IN);
        // 3) edge means, degrees, node scatter
        k_edge_mean<<<ROWS, 256, 0, stream>>>(xprime, nn, edge, N0);
        hipMemsetAsync(cnt, 0, (size_t)ROWS * 4, stream);
        k_count<<<(ROWS + 255) / 256, 256, 0, stream>>>(nn, cnt, ROWS, N0);
        hipMemsetAsync(accv, 0, (size_t)ROWS * HG * 4, stream);
        k_scatter<<<ROWS, 256, 0, stream>>>(edge, nn, accv, N0);
        // 4) hidden = lrelu(acc/Dv + bias); also refresh bf16 cols [0,512)
        k_finalize<<<(ROWS * HG + 255) / 256, 256, 0, stream>>>(
            accv, cnt, bsT[l], hid, hcat, ROWS * HG);

        if (l < 2) {
            const unsigned short* xb = (l == 0) ? x1b : x2b;
            const unsigned short* wt = (l == 0) ? wf1t : wf2t;
            const float* bf = (l == 0) ? bf1 : bf2;
            const float* ci = (l == 0) ? c1 : c2;
            float* ei = (l == 0) ? e1 : e2;
            int Ni = (l == 0) ? N1 : N2;
            int Mi = BATCH * Ni;
            // e_i = x_i @ Wf + bf
            k_gemm<<<dim3((Mi + 15) / 16, H_UP / 16), 32, 0, stream>>>(
                xb, wt, bf, ei, Mi, H_UP, D_IN);
            // nearest upper-level node per level-0 node (2D coords)
            k_nearest<<<(ROWS + 255) / 256, 256, 0, stream>>>(c0, ci, nearest, Ni);
            // hcat[:, 512:1024) = bf16(e_i[nearest])
            k_gather<<<(ROWS * H_UP + 255) / 256, 256, 0, stream>>>(
                ei, nearest, hcat, Ni, ROWS * H_UP);
            // norms of the new 1024-dim concat features
            k_rownorms<<<ROWS, 128, 0, stream>>>(hcat, norms, D_IN);
        }
    }

    // final pooling + last fc
    k_pool<<<(BATCH * HG + 255) / 256, 256, 0, stream>>>(hid, pool);
    k_final<<<(BATCH * TGT + 127) / 128, 128, 0, stream>>>(
        pool, Wlast, blast, (float*)d_out);
}
